// SparseMoe_55611236549112
// MI455X (gfx1250) — compile-verified, matrix operand order checked by reading, not run on hardware
//
#include <hip/hip_runtime.h>

// ---------------- problem constants (match reference) ----------------
#define HID  1024
#define NEXP 8
#define IMO  1024
#define SIMO 2816
#define NTOK 4096   // B*S = 2*2048

// sched_group_barrier masks
#define SG_WMMA      0x008
#define SG_VMEM_READ 0x020
#define SG_DS_READ   0x100
#define SG_DS_WRITE  0x200

// ---------------- WMMA types ----------------
typedef __attribute__((ext_vector_type(16))) __bf16 v16bf;
typedef __attribute__((ext_vector_type(8)))  float  v8f;

union FragU { v16bf v; uint4 q[2]; };

// Global 16x32 bf16 fragment load (A path).
// ISA layout (16-bit A 16x32): lane = 16*h + m; VGPR0-3 hold K=8h..8h+7,
// VGPR4-7 hold K=16+8h..16+8h+7 -> two contiguous 16B chunks 32B apart.
__device__ __forceinline__ v16bf load_frag16(const __bf16* __restrict__ base,
                                             int row, int ld, int k0, int h) {
  const uint4* p = reinterpret_cast<const uint4*>(base + (size_t)row * (size_t)ld + k0 + 8 * h);
  FragU f;
  f.q[0] = p[0];
  f.q[1] = p[2];   // +32 bytes = +16 bf16 elements
  return f.v;
}

// LDS fragment load from a row-major [rows][32] bf16 tile -> ds_load_b128 x2
__device__ __forceinline__ v16bf lds_frag(const __bf16* tile, int row, int h) {
  const uint4* p = reinterpret_cast<const uint4*>(tile + row * 32 + 8 * h);
  FragU f;
  f.q[0] = p[0];
  f.q[1] = p[2];
  return f.v;
}

__device__ __forceinline__ v8f wmma_bf16(v16bf a, v16bf b, v8f c) {
  return __builtin_amdgcn_wmma_f32_16x16x32_bf16(false, a, false, b, (short)0, c, false, false);
}

// ---------------- fp32 -> bf16 convert ----------------
__global__ void cvt_f32_bf16(const float* __restrict__ in, __bf16* __restrict__ out, size_t n) {
  size_t i = ((size_t)blockIdx.x * blockDim.x + threadIdx.x) * 4;
  if (i + 3 < n) {
    float4 v = *reinterpret_cast<const float4*>(in + i);
    out[i + 0] = (__bf16)v.x;
    out[i + 1] = (__bf16)v.y;
    out[i + 2] = (__bf16)v.z;
    out[i + 3] = (__bf16)v.w;
  }
}

// ---------------- router: top2 combine weights + shared sigmoid gate ----------------
__global__ __launch_bounds__(256) void router_kernel(const float* __restrict__ X,
                                                     const float* __restrict__ GW,
                                                     const float* __restrict__ SGW,
                                                     float* __restrict__ combine,
                                                     float* __restrict__ sgate) {
  const int lane = threadIdx.x & 31;
  const int wv   = threadIdx.x >> 5;
  const int t    = blockIdx.x * 8 + wv;
  const float* x = X + (size_t)t * HID;

  float acc[9];
#pragma unroll
  for (int e = 0; e < 9; e++) acc[e] = 0.0f;

  for (int j = 0; j < HID / 32; j++) {
    const int hidx = j * 32 + lane;
    const float xv = x[hidx];
#pragma unroll
    for (int e = 0; e < 8; e++) acc[e] += xv * GW[e * HID + hidx];
    acc[8] += xv * SGW[hidx];
  }
#pragma unroll
  for (int e = 0; e < 9; e++) {
#pragma unroll
    for (int off = 16; off > 0; off >>= 1)
      acc[e] += __shfl_xor(acc[e], off, 32);
  }

  if (lane == 0) {
    float z1 = acc[0]; int i1 = 0;
#pragma unroll
    for (int e = 1; e < 8; e++) if (acc[e] > z1) { z1 = acc[e]; i1 = e; }
    float z2 = -3.4e38f; int i2 = 0;
#pragma unroll
    for (int e = 0; e < 8; e++) if (e != i1 && acc[e] > z2) { z2 = acc[e]; i2 = e; }
    const float v2 = __expf(z2 - z1);   // softmax denom cancels in top-2 ratio
    const float c1 = 1.0f / (1.0f + v2);
    const float c2 = v2 / (1.0f + v2);
#pragma unroll
    for (int e = 0; e < 8; e++)
      combine[(size_t)t * NEXP + e] = (e == i1) ? c1 : ((e == i2) ? c2 : 0.0f);
    sgate[t] = 1.0f / (1.0f + __expf(-acc[8]));
  }
}

// ---------------- gate/up GEMM with fused SiLU(g)*u epilogue ----------------
// grid = (Nact/64, NTOK/128); block = 256 (8 waves); wave tile 16(M) x 64(N) for g and u.
// B tiles (64x32 each) double-buffered in LDS; A fragment register-rotated.
__global__ __launch_bounds__(256) void gateup_kernel(const __bf16* __restrict__ X,
                                                     const __bf16* __restrict__ Wg,
                                                     const __bf16* __restrict__ Wu,
                                                     __bf16* __restrict__ act,
                                                     int K, int Nact) {
  __shared__ __bf16 lG[2][64 * 32];
  __shared__ __bf16 lU[2][64 * 32];

  const int tid  = threadIdx.x;
  const int lane = tid & 31;
  const int wv   = tid >> 5;
  const int m    = lane & 15;
  const int h    = lane >> 4;
  const int rowBase = blockIdx.y * 128 + wv * 16;
  const int colBase = blockIdx.x * 64;

  // staging: 256 threads, 64 rows, 4 x 16B chunks per 64B row, per matrix
  const int srow = tid >> 2;
  const int sq   = tid & 3;
  const __bf16* gG = Wg + (size_t)(colBase + srow) * K + 8 * sq;
  const __bf16* gU = Wu + (size_t)(colBase + srow) * K + 8 * sq;
  const int ldsOff = srow * 32 + 8 * sq;

  v8f accG[4], accU[4];
#pragma unroll
  for (int i = 0; i < 4; i++) { accG[i] = (v8f){}; accU[i] = (v8f){}; }

  // prologue: tile 0 -> buffer 0; first A fragment
  {
    uint4 rg = *reinterpret_cast<const uint4*>(gG);
    uint4 ru = *reinterpret_cast<const uint4*>(gU);
    *reinterpret_cast<uint4*>(&lG[0][ldsOff]) = rg;
    *reinterpret_cast<uint4*>(&lU[0][ldsOff]) = ru;
  }
  v16bf a_cur = load_frag16(X, rowBase + m, K, 0, h);
  __syncthreads();

  const int nk = K >> 5;
  for (int kt = 0; kt < nk - 1; kt++) {   // branch-free main body
    const int cur = kt & 1;
    // prefetch next B tiles + next A fragment (waits land an iteration later)
    uint4 rg = *reinterpret_cast<const uint4*>(gG + (kt + 1) * 32);
    uint4 ru = *reinterpret_cast<const uint4*>(gU + (kt + 1) * 32);
    v16bf a_nxt = load_frag16(X, rowBase + m, K, (kt + 1) * 32, h);

    v16bf bg[4], bu[4];
#pragma unroll
    for (int nb = 0; nb < 4; nb++) bg[nb] = lds_frag(lG[cur], nb * 16 + m, h);
#pragma unroll
    for (int nb = 0; nb < 4; nb++) bu[nb] = lds_frag(lU[cur], nb * 16 + m, h);
#pragma unroll
    for (int nb = 0; nb < 4; nb++) accG[nb] = wmma_bf16(a_cur, bg[nb], accG[nb]);
#pragma unroll
    for (int nb = 0; nb < 4; nb++) accU[nb] = wmma_bf16(a_cur, bu[nb], accU[nb]);

    *reinterpret_cast<uint4*>(&lG[cur ^ 1][ldsOff]) = rg;
    *reinterpret_cast<uint4*>(&lU[cur ^ 1][ldsOff]) = ru;

    // pipeline shape: vmem reads -> ds reads -> wmma block -> ds writes
    __builtin_amdgcn_sched_group_barrier(SG_VMEM_READ, 4, 0);
    __builtin_amdgcn_sched_group_barrier(SG_DS_READ, 16, 0);
    __builtin_amdgcn_sched_group_barrier(SG_WMMA, 8, 0);
    __builtin_amdgcn_sched_group_barrier(SG_DS_WRITE, 2, 0);

    __syncthreads();
    a_cur = a_nxt;
  }
  {   // tail iteration (no prefetch/staging)
    const int cur = (nk - 1) & 1;
    v16bf bg[4], bu[4];
#pragma unroll
    for (int nb = 0; nb < 4; nb++) bg[nb] = lds_frag(lG[cur], nb * 16 + m, h);
#pragma unroll
    for (int nb = 0; nb < 4; nb++) bu[nb] = lds_frag(lU[cur], nb * 16 + m, h);
#pragma unroll
    for (int nb = 0; nb < 4; nb++) accG[nb] = wmma_bf16(a_cur, bg[nb], accG[nb]);
#pragma unroll
    for (int nb = 0; nb < 4; nb++) accU[nb] = wmma_bf16(a_cur, bu[nb], accU[nb]);
  }

  // C/D layout: element (M = r + 8*h, N = lane&15) lives in VGPR r
#pragma unroll
  for (int nb = 0; nb < 4; nb++) {
    const int col = colBase + nb * 16 + m;
#pragma unroll
    for (int r = 0; r < 8; r++) {
      const int row = rowBase + h * 8 + r;
      const float g = accG[nb][r];
      const float u = accU[nb][r];
      const float s = g / (1.0f + __expf(-g));   // SiLU
      act[(size_t)row * Nact + col] = (__bf16)(s * u);
    }
  }
}

// ---------------- down GEMM: out[t,h] (+)= scale[t] * (act @ Wd^T)[t,h] ----------------
// grid = (HID/128, NTOK/128); block = 256; wave tile 16(M) x 128(N).
// B tile (128x32) double-buffered in LDS; A fragment register-rotated.
__global__ __launch_bounds__(256) void down_kernel(const __bf16* __restrict__ A,
                                                   const __bf16* __restrict__ W,
                                                   const float* __restrict__ scale,
                                                   int sstride,
                                                   float* __restrict__ out,
                                                   int K, int accumulate) {
  __shared__ __bf16 lB[2][128 * 32];

  const int tid  = threadIdx.x;
  const int lane = tid & 31;
  const int wv   = tid >> 5;
  const int m    = lane & 15;
  const int h    = lane >> 4;
  const int rowBase = blockIdx.y * 128 + wv * 16;
  const int colBase = blockIdx.x * 128;

  // staging: 256 threads, 128 rows, 2 x 32B halves per 64B row
  const int srow  = tid >> 1;
  const int shalf = tid & 1;
  const __bf16* gB = W + (size_t)(colBase + srow) * K + 16 * shalf;
  const int ldsOff = srow * 32 + 16 * shalf;

  v8f acc[8];
#pragma unroll
  for (int i = 0; i < 8; i++) acc[i] = (v8f){};

  {
    const uint4* g = reinterpret_cast<const uint4*>(gB);
    uint4 r0 = g[0], r1 = g[1];
    *reinterpret_cast<uint4*>(&lB[0][ldsOff])     = r0;
    *reinterpret_cast<uint4*>(&lB[0][ldsOff + 8]) = r1;
  }
  v16bf a_cur = load_frag16(A, rowBase + m, K, 0, h);
  __syncthreads();

  const int nk = K >> 5;
  for (int kt = 0; kt < nk - 1; kt++) {   // branch-free main body
    const int cur = kt & 1;
    const uint4* g = reinterpret_cast<const uint4*>(gB + (kt + 1) * 32);
    uint4 r0 = g[0], r1 = g[1];
    v16bf a_nxt = load_frag16(A, rowBase + m, K, (kt + 1) * 32, h);

    v16bf bf[8];
#pragma unroll
    for (int nb = 0; nb < 8; nb++) bf[nb] = lds_frag(lB[cur], nb * 16 + m, h);
#pragma unroll
    for (int nb = 0; nb < 8; nb++) acc[nb] = wmma_bf16(a_cur, bf[nb], acc[nb]);

    *reinterpret_cast<uint4*>(&lB[cur ^ 1][ldsOff])     = r0;
    *reinterpret_cast<uint4*>(&lB[cur ^ 1][ldsOff + 8]) = r1;

    __builtin_amdgcn_sched_group_barrier(SG_VMEM_READ, 4, 0);
    __builtin_amdgcn_sched_group_barrier(SG_DS_READ, 16, 0);
    __builtin_amdgcn_sched_group_barrier(SG_WMMA, 8, 0);
    __builtin_amdgcn_sched_group_barrier(SG_DS_WRITE, 2, 0);

    __syncthreads();
    a_cur = a_nxt;
  }
  {   // tail iteration
    const int cur = (nk - 1) & 1;
    v16bf bf[8];
#pragma unroll
    for (int nb = 0; nb < 8; nb++) bf[nb] = lds_frag(lB[cur], nb * 16 + m, h);
#pragma unroll
    for (int nb = 0; nb < 8; nb++) acc[nb] = wmma_bf16(a_cur, bf[nb], acc[nb]);
  }

#pragma unroll
  for (int nb = 0; nb < 8; nb++) {
    const int col = colBase + nb * 16 + m;
#pragma unroll
    for (int r = 0; r < 8; r++) {
      const int t = rowBase + h * 8 + r;
      const float s = scale[(size_t)t * sstride];
      const size_t idx = (size_t)t * HID + col;
      float v = s * acc[nb][r];
      if (accumulate) v += out[idx];
      out[idx] = v;
    }
  }
}

// ---------------- host launcher ----------------
extern "C" void kernel_launch(void* const* d_in, const int* in_sizes, int n_in,
                              void* d_out, int out_size, void* d_ws, size_t ws_size,
                              hipStream_t stream) {
  const float* hx  = (const float*)d_in[0];  // [T,H]
  const float* gw  = (const float*)d_in[1];  // [E,H]
  const float* wg  = (const float*)d_in[2];  // [E,I,H]
  const float* wu  = (const float*)d_in[3];  // [E,I,H]
  const float* wd  = (const float*)d_in[4];  // [E,H,I]
  const float* swg = (const float*)d_in[5];  // [SI,H]
  const float* swu = (const float*)d_in[6];  // [SI,H]
  const float* swd = (const float*)d_in[7];  // [H,SI]
  const float* sgw = (const float*)d_in[8];  // [1,H]
  float* out = (float*)d_out;                // [T,H]
  (void)in_sizes; (void)n_in; (void)out_size; (void)ws_size;

  char* ws = (char*)d_ws;
  size_t off = 0;
  auto take = [&](size_t bytes) -> void* {
    void* p = (void*)(ws + off);
    off += (bytes + 255) & ~(size_t)255;
    return p;
  };
  __bf16* Xb   = (__bf16*)take((size_t)NTOK * HID * 2);
  __bf16* Wgb  = (__bf16*)take((size_t)NEXP * IMO * HID * 2);
  __bf16* Wub  = (__bf16*)take((size_t)NEXP * IMO * HID * 2);
  __bf16* Wdb  = (__bf16*)take((size_t)NEXP * HID * IMO * 2);
  __bf16* Swgb = (__bf16*)take((size_t)SIMO * HID * 2);
  __bf16* Swub = (__bf16*)take((size_t)SIMO * HID * 2);
  __bf16* Swdb = (__bf16*)take((size_t)HID * SIMO * 2);
  float*  combine = (float*)take((size_t)NTOK * NEXP * 4);
  float*  sgate   = (float*)take((size_t)NTOK * 4);
  __bf16* act  = (__bf16*)take((size_t)NTOK * SIMO * 2);

  auto cvt = [&](const float* src, __bf16* dst, size_t n) {
    const int threads = 256;
    const int blocks  = (int)((n / 4 + threads - 1) / threads);
    cvt_f32_bf16<<<blocks, threads, 0, stream>>>(src, dst, n);
  };
  cvt(hx,  Xb,   (size_t)NTOK * HID);
  cvt(wg,  Wgb,  (size_t)NEXP * IMO * HID);
  cvt(wu,  Wub,  (size_t)NEXP * IMO * HID);
  cvt(wd,  Wdb,  (size_t)NEXP * HID * IMO);
  cvt(swg, Swgb, (size_t)SIMO * HID);
  cvt(swu, Swub, (size_t)SIMO * HID);
  cvt(swd, Swdb, (size_t)HID * SIMO);

  router_kernel<<<NTOK / 8, 256, 0, stream>>>(hx, gw, sgw, combine, sgate);

  // shared expert: out = sgate * (silu(X@swg^T) * (X@swu^T)) @ swd^T
  gateup_kernel<<<dim3(SIMO / 64, NTOK / 128), 256, 0, stream>>>(Xb, Swgb, Swub, act, HID, SIMO);
  down_kernel<<<dim3(HID / 128, NTOK / 128), 256, 0, stream>>>(act, Swdb, sgate, 1, out, SIMO, 0);

  // routed experts: out += combine[:,e] * (silu(X@wg_e^T) * (X@wu_e^T)) @ wd_e^T
  for (int e = 0; e < NEXP; e++) {
    gateup_kernel<<<dim3(IMO / 64, NTOK / 128), 256, 0, stream>>>(
        Xb, Wgb + (size_t)e * IMO * HID, Wub + (size_t)e * IMO * HID, act, HID, IMO);
    down_kernel<<<dim3(HID / 128, NTOK / 128), 256, 0, stream>>>(
        act, Wdb + (size_t)e * HID * IMO, combine + e, NEXP, out, IMO, 1);
  }
}
